// TurboQuantKV_69114613730478
// MI455X (gfx1250) — matrix operand order repaired and checked by reading, not compiled, not run on hardware
//
#include <hip/hip_runtime.h>

// ---------------------------------------------------------------------------
// TurboQuant KV fused pipeline for MI455X (gfx1250, wave32, WMMA).
//
//  Pass 1 (pack):  R,P -> bf16 in WMMA-B fragment-major layout in d_ws
//                  kv,query -> bf16 row-major in d_ws
//  Pass 2 (fused): kv@R (pass A), q@R (pass B) with v_wmma_f32_16x16x32_bf16,
//                  quantize from f32 accs -> error/kv_rot -> LDS,
//                  dot_pq vs LDS, q_rot overwrites kv_rot,
//                  err@P + q_rot@P, sign/scale reductions -> out.
//
// 512-thread workgroups (16 wave32s): 4 waves/SIMD -> 256-VGPR budget/wave,
// so the 128-VGPR accumulator set (16 tiles) stays register-resident (the
// 1024-thread version was hard-capped at 128 VGPRs and spilled to scratch).
// d_ws requirement: 32MB (Rbf) + 8MB (Pbf) + 64MB (kvbf) + 64MB (qbf) = 168MB.
// ---------------------------------------------------------------------------

#define DIM    4096
#define JLDIM  1024
#define NBATCH 2
#define SEQ    4096
#define MROWS  16
#define NWAVES 16
#define NT1    16  // GEMM1 pass: N-tiles per wave (16*16 = 256 cols of D)
#define NT2    4   // GEMM2/3: N-tiles per wave (4*16 = 64 cols of JL)
#define KCH    32  // K per WMMA

typedef __bf16 bf16;
typedef __attribute__((ext_vector_type(16))) __bf16 v16bf;
typedef __attribute__((ext_vector_type(8)))  __bf16 v8bf;
typedef __attribute__((ext_vector_type(8)))  float  v8f;

union AF { v16bf v; v8bf h[2]; };

__device__ __forceinline__ v8f wmma_bf16(const v16bf& a, const v16bf& b, v8f c) {
  return __builtin_amdgcn_wmma_f32_16x16x32_bf16(
      false, a, false, b, (short)0, c, false, false);
}

__device__ __forceinline__ v8f v8f_zero() {
  v8f z;
#pragma unroll
  for (int i = 0; i < 8; ++i) z[i] = 0.0f;
  return z;
}

// ---------------------------------------------------------------------------
// Pack a K x ncols f32 matrix into bf16 "B-fragment-major" layout:
//   dst[((k/32)*ncols + n)*32 + (k%32)] = (bf16) src[k*ncols + n]
// ---------------------------------------------------------------------------
extern "C" __global__ __launch_bounds__(256)
void pack_b_bf16(const float* __restrict__ src, bf16* __restrict__ dst, int ncols) {
  const int gid = blockIdx.x * blockDim.x + threadIdx.x;  // one per (kc, n)
  const int n   = gid % ncols;
  const int kc  = gid / ncols;
  union { bf16 e[KCH]; v8bf h[4]; } buf;
#pragma unroll
  for (int kin = 0; kin < KCH; ++kin)
    buf.e[kin] = (bf16)src[(size_t)(kc * KCH + kin) * ncols + n];
  v8bf* d = (v8bf*)&dst[((size_t)kc * ncols + n) * KCH];
#pragma unroll
  for (int i = 0; i < 4; ++i) d[i] = buf.h[i];
}

// Plain elementwise f32 -> bf16 (row-major preserved), 4 elems per thread.
extern "C" __global__ __launch_bounds__(256)
void pack_a_bf16(const float* __restrict__ src, bf16* __restrict__ dst) {
  const size_t gid = ((size_t)blockIdx.x * blockDim.x + threadIdx.x) * 4;
  const float4 v = *(const float4*)&src[gid];
  union { bf16 o[4]; unsigned long long u; } pk;
  pk.o[0] = (bf16)v.x; pk.o[1] = (bf16)v.y; pk.o[2] = (bf16)v.z; pk.o[3] = (bf16)v.w;
  *(unsigned long long*)&dst[gid] = pk.u;
}

// ---------------------------------------------------------------------------
// One full-K rotation pass: acc[t] += A_blk(16xK) @ R(:, n0+256 cols), bf16.
// Barrier-free: A frags are 2x b128 (1KB/chunk shared by all waves via L2/WGP$),
// B frags are 2x contiguous b128 from the fragment-major packed matrix.
// ---------------------------------------------------------------------------
__device__ __forceinline__ void rot_pass(const bf16* __restrict__ Abf,
                                         const bf16* __restrict__ Bfm,
                                         size_t rowbase, int n0,
                                         int l16, int half, int lane,
                                         v8f acc[NT1]) {
  for (int kc = 0; kc < DIM / KCH; ++kc) {
    AF fa;  // lane -> row l16; K groups 8*half..+7 and 16+8*half..+7
    const size_t aoff = (rowbase + l16) * DIM + kc * KCH + 8 * half;
    fa.h[0] = *(const v8bf*)&Abf[aoff];
    fa.h[1] = *(const v8bf*)&Abf[aoff + 16];

    if (kc + 1 < DIM / KCH)  // stream next B slab toward L2 -> global_prefetch_b8
      __builtin_prefetch(&Bfm[(((size_t)(kc + 1) * DIM) + n0 + lane * 8) * KCH], 0, 1);

    const bf16* rbase = &Bfm[((size_t)kc * DIM) * KCH + 8 * half];
#pragma unroll
    for (int t = 0; t < NT1; ++t) {
      const v8bf* bp = (const v8bf*)&rbase[(size_t)(n0 + t * 16 + l16) * KCH];
      AF fb;  // lane -> column; K groups 8*half..+7 and 16+8*half..+7
      fb.h[0] = bp[0];
      fb.h[1] = bp[2];
      acc[t] = wmma_bf16(fa.v, fb.v, acc[t]);
    }
  }
}

// ---------------------------------------------------------------------------
// Fused main kernel: one workgroup (16 waves, 512 thr) per block of 16 rows.
// ---------------------------------------------------------------------------
extern "C" __global__ __launch_bounds__(NWAVES * 32, 1)
void turboquant_fused(const bf16* __restrict__ kvbf,
                      const bf16* __restrict__ qbf,
                      const bf16* __restrict__ Rbf,   // fragment-major
                      const bf16* __restrict__ Pbf,   // fragment-major
                      const float* __restrict__ noise,
                      float* __restrict__ out)
{
  // Dynamic LDS: buf0 = error[16][4096], buf1 = kv_rot (later q_rot) -> 256 KB
  extern __shared__ char dynsmem[];
  bf16* err_lds = (bf16*)dynsmem;
  bf16* kq_lds  = err_lds + MROWS * DIM;   // kv_rot in pass1, q_rot afterwards

  __shared__ float s_norm2 [MROWS];
  __shared__ float s_dotpq [MROWS];
  __shared__ float s_dotqjl[MROWS];
  __shared__ int   s_scale [MROWS];        // f32 bits of row max |err_proj| (>=0)

  const int tid  = threadIdx.x;
  const int wid  = tid >> 5;
  const int lane = tid & 31;
  const int half = lane >> 4;              // 0: lanes 0-15, 1: lanes 16-31
  const int l16  = lane & 15;

  const int blk = blockIdx.x;
  const int bb  = blk / (SEQ / MROWS);
  const int s0  = (blk % (SEQ / MROWS)) * MROWS;
  const size_t rowbase = (size_t)(bb * SEQ + s0);

  if (tid < MROWS) {
    s_norm2[tid] = 0.0f; s_dotpq[tid] = 0.0f; s_dotqjl[tid] = 0.0f; s_scale[tid] = 0;
  }
  __syncthreads();

  const int n0 = wid * (NT1 * 16);         // this wave's 256 columns of D
  v8f acc[NT1];

  // ============================ Pass A: kv @ R ==========================
#pragma unroll
  for (int t = 0; t < NT1; ++t) acc[t] = v8f_zero();
  rot_pass(kvbf, Rbf, rowbase, n0, l16, half, lane, acc);

  // ---- row sum-of-squares from exact f32 accumulators
  {
    float pn[8];
#pragma unroll
    for (int r = 0; r < 8; ++r) pn[r] = 0.0f;
#pragma unroll
    for (int t = 0; t < NT1; ++t)
#pragma unroll
      for (int r = 0; r < 8; ++r) { float c = acc[t][r]; pn[r] += c * c; }
#pragma unroll
    for (int r = 0; r < 8; ++r)
      for (int m = 1; m < 16; m <<= 1) pn[r] += __shfl_xor(pn[r], m);
    if (l16 == 0)
#pragma unroll
      for (int r = 0; r < 8; ++r) atomicAdd(&s_norm2[half * 8 + r], pn[r]);
  }
  __syncthreads();

  // ---- quantize from f32 accs; stash error and kv_rot (bf16) in LDS
  {
    float nrm[8];
#pragma unroll
    for (int r = 0; r < 8; ++r) nrm[r] = sqrtf(s_norm2[half * 8 + r]);
#pragma unroll
    for (int t = 0; t < NT1; ++t) {
      const int col = n0 + t * 16 + l16;
#pragma unroll
      for (int r = 0; r < 8; ++r) {
        const int row = half * 8 + r;
        const float c   = acc[t][r];
        const float dir = c / (nrm[r] + 1e-8f);
        const float nz  = fminf(fmaxf(dir * (64.0f / 3.0f), -1.0f), 1.0f);
        const float qn  = rintf(nz * 7.0f) * (1.0f / 7.0f);      // RNE = jnp.round
        const float khat = qn * (3.0f / 64.0f) * nrm[r];
        err_lds[row * DIM + col] = (bf16)(c - khat);
        kq_lds [row * DIM + col] = (bf16)c;                       // kv_rot
      }
    }
  }
  __syncthreads();

  // ============================ Pass B: q @ R ===========================
#pragma unroll
  for (int t = 0; t < NT1; ++t) acc[t] = v8f_zero();
  rot_pass(qbf, Rbf, rowbase, n0, l16, half, lane, acc);

  // ---- dot_pq = sum q_rot * (kv_rot - error); q_rot replaces kv_rot in LDS
  {
    float pd[8];
#pragma unroll
    for (int r = 0; r < 8; ++r) pd[r] = 0.0f;
#pragma unroll
    for (int t = 0; t < NT1; ++t) {
      const int col = n0 + t * 16 + l16;
#pragma unroll
      for (int r = 0; r < 8; ++r) {
        const int row = half * 8 + r;
        const float qv  = acc[t][r];
        const float kvr = (float)kq_lds [row * DIM + col];
        const float er  = (float)err_lds[row * DIM + col];
        pd[r] += qv * (kvr - er);                                 // q . k_hat
        kq_lds[row * DIM + col] = (bf16)qv;                       // -> q_rot
      }
    }
#pragma unroll
    for (int r = 0; r < 8; ++r)
      for (int m = 1; m < 16; m <<= 1) pd[r] += __shfl_xor(pd[r], m);
    if (l16 == 0)
#pragma unroll
      for (int r = 0; r < 8; ++r) atomicAdd(&s_dotpq[half * 8 + r], pd[r]);
  }
  __syncthreads();   // err/q_rot LDS stable for all waves

  // =====================================================================
  // GEMM2/3: error_proj = err @ P, q_proj = q_rot @ P (64 JL cols / wave).
  // A frags: 16B-aligned ds_read_b128 pairs; B frags: contiguous b128 Pbf.
  // =====================================================================
  const int jn0 = wid * (NT2 * 16);
  v8f ep[NT2], qp[NT2];
#pragma unroll
  for (int t = 0; t < NT2; ++t) { ep[t] = v8f_zero(); qp[t] = v8f_zero(); }

  for (int kc = 0; kc < DIM / KCH; ++kc) {
    const int kk = kc * KCH;
    AF fe, fqr;
    const int o0 = l16 * DIM + kk + 8 * half;
    fe.h[0]  = *(const v8bf*)&err_lds[o0];
    fe.h[1]  = *(const v8bf*)&err_lds[o0 + 16];
    fqr.h[0] = *(const v8bf*)&kq_lds[o0];
    fqr.h[1] = *(const v8bf*)&kq_lds[o0 + 16];
    const bf16* pbase = &Pbf[((size_t)kc * JLDIM) * KCH + 8 * half];
#pragma unroll
    for (int t = 0; t < NT2; ++t) {
      const int col = jn0 + t * 16 + l16;
      const v8bf* bp = (const v8bf*)&pbase[(size_t)col * KCH];
      AF fb;
      fb.h[0] = bp[0];
      fb.h[1] = bp[2];
      ep[t] = wmma_bf16(fe.v,  fb.v, ep[t]);  // B fragment shared
      qp[t] = wmma_bf16(fqr.v, fb.v, qp[t]);
    }
  }

  // ---- error_scale = max |error_proj| per row (int atomicMax: vals >= 0)
  {
    float pm[8];
#pragma unroll
    for (int r = 0; r < 8; ++r) pm[r] = 0.0f;
#pragma unroll
    for (int t = 0; t < NT2; ++t)
#pragma unroll
      for (int r = 0; r < 8; ++r) pm[r] = fmaxf(pm[r], fabsf(ep[t][r]));
#pragma unroll
    for (int r = 0; r < 8; ++r)
      for (int m = 1; m < 16; m <<= 1) pm[r] = fmaxf(pm[r], __shfl_xor(pm[r], m));
    if (l16 == 0)
#pragma unroll
      for (int r = 0; r < 8; ++r)
        atomicMax(&s_scale[half * 8 + r], __float_as_int(pm[r]));
  }
  __syncthreads();

  // ---- dot_qjl = sum q_proj * sign(err_n - rand_noise) * scale
  {
    float pj[8];
#pragma unroll
    for (int r = 0; r < 8; ++r) pj[r] = 0.0f;
#pragma unroll
    for (int t = 0; t < NT2; ++t) {
      const int col = jn0 + t * 16 + l16;
#pragma unroll
      for (int r = 0; r < 8; ++r) {
        const int row = half * 8 + r;
        const float scale = fmaxf(__int_as_float(s_scale[row]), 1e-8f);
        const float en = ep[t][r] / scale;
        const float rn = noise[(rowbase + row) * JLDIM + col] * 2.0f - 1.0f;
        const float d  = en - rn;
        const float sg = (d > 0.0f) ? 1.0f : ((d < 0.0f) ? -1.0f : 0.0f);
        pj[r] += qp[t][r] * sg * scale;
      }
    }
#pragma unroll
    for (int r = 0; r < 8; ++r)
      for (int m = 1; m < 16; m <<= 1) pj[r] += __shfl_xor(pj[r], m);
    if (l16 == 0)
#pragma unroll
      for (int r = 0; r < 8; ++r) atomicAdd(&s_dotqjl[half * 8 + r], pj[r]);
  }
  __syncthreads();

  if (tid < MROWS)
    out[rowbase + tid] = s_dotpq[tid] + s_dotqjl[tid];
}

// ---------------------------------------------------------------------------
extern "C" void kernel_launch(void* const* d_in, const int* in_sizes, int n_in,
                              void* d_out, int out_size, void* d_ws, size_t ws_size,
                              hipStream_t stream) {
  (void)in_sizes; (void)n_in; (void)ws_size; (void)out_size;
  const float* kv    = (const float*)d_in[0];
  const float* query = (const float*)d_in[1];
  const float* Rm    = (const float*)d_in[2];
  const float* Pm    = (const float*)d_in[3];
  const float* noise = (const float*)d_in[4];
  float* out = (float*)d_out;

  // Workspace carve (168 MB total, 16B-aligned blocks).
  bf16* Rbf  = (bf16*)d_ws;                            //  32 MB
  bf16* Pbf  = Rbf  + (size_t)DIM * DIM;               //   8 MB
  bf16* kvbf = Pbf  + (size_t)DIM * JLDIM;             //  64 MB
  bf16* qbf  = kvbf + (size_t)NBATCH * SEQ * DIM;      //  64 MB

  // ---- pass 1: precision/layout packing -------------------------------
  hipLaunchKernelGGL(pack_b_bf16, dim3((DIM / KCH) * DIM   / 256), dim3(256), 0, stream,
                     Rm, Rbf, DIM);
  hipLaunchKernelGGL(pack_b_bf16, dim3((DIM / KCH) * JLDIM / 256), dim3(256), 0, stream,
                     Pm, Pbf, JLDIM);
  const size_t na = (size_t)NBATCH * SEQ * DIM;        // 33.5M elems each
  hipLaunchKernelGGL(pack_a_bf16, dim3((unsigned)(na / 4 / 256)), dim3(256), 0, stream,
                     kv, kvbf);
  hipLaunchKernelGGL(pack_a_bf16, dim3((unsigned)(na / 4 / 256)), dim3(256), 0, stream,
                     query, qbf);

  // ---- pass 2: fused WMMA pipeline ------------------------------------
  const size_t smem = (size_t)2 * MROWS * DIM * sizeof(bf16);  // 256 KB dynamic LDS
  dim3 grid((NBATCH * SEQ) / MROWS);   // 512 workgroups
  dim3 block(NWAVES * 32);             // 512 threads = 16 wave32s
  hipLaunchKernelGGL(turboquant_fused, grid, block, smem, stream,
                     kvbf, qbf, Rbf, Pbf, noise, out);
}
